// LSTMModel_15702400434245
// MI455X (gfx1250) — matrix operand order repaired and checked by reading, not compiled, not run on hardware
//
#include <hip/hip_runtime.h>

// CDNA5 (gfx1250) fully-fused 2-layer LSTM, wave32, f16 WMMA 16x16x32.
// One wave per 16-row batch tile; weights resident in VGPRs; h transposed
// D->A through LDS (f16) each step; x prefetched one step ahead; biases
// folded into the activation exp2 FMAs; no intermediate tensors touch HBM.

typedef __attribute__((ext_vector_type(16))) _Float16 v16h;
typedef __attribute__((ext_vector_type(8)))  _Float16 v8h;
typedef __attribute__((ext_vector_type(8)))  float    v8f;
typedef __attribute__((ext_vector_type(4)))  float    v4f;

#define HID 32      // hidden size
#define SEQ 512     // timesteps
#define NT  8       // 128 gate outputs / 16-wide N tiles
#define L2E 1.4426950408889634f

// sigmoid(x + b) with pre-scaled bias c = -b*log2(e): 1 fma + v_exp + add + v_rcp
static __device__ __forceinline__ float sigm_b(float x, float c) {
  float e = __builtin_amdgcn_exp2f(__builtin_fmaf(x, -L2E, c));
  return __builtin_amdgcn_rcpf(1.0f + e);
}
// tanh(x + b) with pre-scaled bias c = -2*b*log2(e)
static __device__ __forceinline__ float tanh_b(float x, float c) {
  float e = __builtin_amdgcn_exp2f(__builtin_fmaf(x, -2.0f * L2E, c));
  float r = __builtin_amdgcn_rcpf(1.0f + e);
  return __builtin_fmaf(2.0f, r, -1.0f);
}
static __device__ __forceinline__ float tanh_p(float x) { return tanh_b(x, 0.0f); }

static __device__ __forceinline__ v8f wmma16(v16h a, v16h b, v8f c) {
  // D = A(16x32 f16) * B(32x16 f16) + C(16x16 f32)
  return __builtin_amdgcn_wmma_f32_16x16x32_f16(
      /*neg_a=*/false, a, /*neg_b=*/false, b,
      /*c_mod=*/(short)0, c, /*reuse_a=*/false, /*reuse_b=*/false);
}

// Pack one 32x16 B-tile of W^T (W is [128][32] row-major) into WMMA B layout:
// lane holds N = lane&15, K-half selected by lane>>4; element e -> K = (e<8?0:16) + 8*ksel + (e&7).
static __device__ __forceinline__ v16h load_btile(const float* __restrict__ W,
                                                  int nt, int n, int ksel) {
  const v4f* p = (const v4f*)(W + (nt * 16 + n) * HID + ksel * 8);
  v4f q0 = p[0], q1 = p[1], q2 = p[4], q3 = p[5];
  v16h b;
#pragma unroll
  for (int e = 0; e < 4; e++) {
    b[e]      = (_Float16)q0[e];
    b[4 + e]  = (_Float16)q1[e];
    b[8 + e]  = (_Float16)q2[e];
    b[12 + e] = (_Float16)q3[e];
  }
  return b;
}

struct LayerState {
  v16h hA;            // recurrent h, A layout (16x32 f16)
  v8f  ca, cb;        // cell state, D layout (n=0..15 / n=16..31)
  v8f  hDa, hDb;      // last h, D layout (kept for final projection)
};

static __device__ __forceinline__ void zero_state(LayerState& S) {
#pragma unroll
  for (int e = 0; e < 16; e++) S.hA[e] = (_Float16)0.0f;
#pragma unroll
  for (int r = 0; r < 8; r++) {
    S.ca[r] = 0.0f; S.cb[r] = 0.0f; S.hDa[r] = 0.0f; S.hDb[r] = 0.0f;
  }
}

// One LSTM cell step for a 16-row batch tile.
// z = Ain @ Wih^T + h @ Whh^T + bias  (torch gate order i,f,g,o; bias folded
// into the activations via pre-scaled constants bs[]).
static __device__ __forceinline__ void lstm_step(
    v16h Ain, const v16h* __restrict__ Bih, const v16h* __restrict__ Bhh,
    const float* __restrict__ bs, const v8f zc, LayerState& S,
    _Float16* shbuf, int m, int ksel) {
  v8f acc[NT];
  // recurrent GEMM first (same A operand across all 8 tiles), C starts at 0
#pragma unroll
  for (int nt = 0; nt < NT; nt++) acc[nt] = wmma16(S.hA, Bhh[nt], zc);
  // input GEMM chained on accumulator
#pragma unroll
  for (int nt = 0; nt < NT; nt++) acc[nt] = wmma16(Ain, Bih[nt], acc[nt]);

  // gates: tiles 0,1=i  2,3=f  4,5=g  6,7=o  (a: n 0..15, b: n 16..31)
#pragma unroll
  for (int r = 0; r < 8; r++) {
    float ia = sigm_b(acc[0][r], bs[0]), fa = sigm_b(acc[2][r], bs[2]);
    float ga = tanh_b(acc[4][r], bs[4]), oa = sigm_b(acc[6][r], bs[6]);
    S.ca[r]  = __builtin_fmaf(fa, S.ca[r], ia * ga);
    S.hDa[r] = oa * tanh_p(S.ca[r]);
    float ib = sigm_b(acc[1][r], bs[1]), fb = sigm_b(acc[3][r], bs[3]);
    float gb = tanh_b(acc[5][r], bs[5]), ob = sigm_b(acc[7][r], bs[7]);
    S.cb[r]  = __builtin_fmaf(fb, S.cb[r], ib * gb);
    S.hDb[r] = ob * tanh_p(S.cb[r]);
  }

  // D -> A transpose via LDS in f16 (single wave: dscnt wait only, no barrier).
#pragma unroll
  for (int r = 0; r < 8; r++) {
    int row = (r + 8 * ksel) * HID;
    shbuf[row + m]      = (_Float16)S.hDa[r];
    shbuf[row + 16 + m] = (_Float16)S.hDb[r];
  }
  asm volatile("s_wait_dscnt 0" ::: "memory");
  // A layout: lane holds row M=lane&15, K in {8k..8k+7, 16+8k..16+8k+7}, k=ksel.
  // Two b128 loads land exactly in the v16h register halves (no per-elem cvt).
  const v8h* p = (const v8h*)(shbuf + m * HID);
  v8h q0 = p[ksel], q1 = p[2 + ksel];
#pragma unroll
  for (int e = 0; e < 8; e++) {
    S.hA[e]     = q0[e];
    S.hA[8 + e] = q1[e];
  }
}

__global__ __launch_bounds__(32) void lstm2_fused(
    const float* __restrict__ x,
    const float* __restrict__ W1,   const float* __restrict__ b1,
    const float* __restrict__ Wih0, const float* __restrict__ Whh0,
    const float* __restrict__ bih0, const float* __restrict__ bhh0,
    const float* __restrict__ Wih1, const float* __restrict__ Whh1,
    const float* __restrict__ bih1, const float* __restrict__ bhh1,
    const float* __restrict__ W2,   const float* __restrict__ b2,
    float* __restrict__ out) {
  __shared__ float sh[16 * 64];  // 4KB: two f16 h-staging buffers; 16x64 f32 at end

  const int lane = threadIdx.x;
  const int m    = lane & 15;   // batch row within tile / N within B-tile
  const int ksel = lane >> 4;   // K-half select
  const int tile = blockIdx.x;
  const float* xr = x + (size_t)(tile * 16 + m) * SEQ;  // I==1: x[b][t]

  _Float16* sh0 = (_Float16*)sh;        // layer-0 staging: 512 halves (1KB)
  _Float16* sh1 = (_Float16*)sh + 512;  // layer-1 staging

  // Per-lane linear_1 weights in A-layout K order: K(e) = (e<8?0:16) + 8*ksel + (e&7)
  float W1k[16], b1k[16];
#pragma unroll
  for (int e = 0; e < 16; e++) {
    int K = ((e < 8) ? 0 : 16) + 8 * ksel + (e & 7);
    W1k[e] = W1[K];
    b1k[e] = b1[K];
  }

  // Resident weight B-tiles (4 matrices x 8 tiles x 8 VGPRs = 256 VGPRs) and
  // biases pre-scaled for folding into the activation exp2 FMAs.
  v16h Bih0v[NT], Bhh0v[NT], Bih1v[NT], Bhh1v[NT];
  float bs0[NT], bs1[NT];
#pragma unroll
  for (int nt = 0; nt < NT; nt++) {
    Bih0v[nt] = load_btile(Wih0, nt, m, ksel);
    Bhh0v[nt] = load_btile(Whh0, nt, m, ksel);
    Bih1v[nt] = load_btile(Wih1, nt, m, ksel);
    Bhh1v[nt] = load_btile(Whh1, nt, m, ksel);
    int g = nt * 16 + m;
    float scale = (nt == 4 || nt == 5) ? (-2.0f * L2E) : (-L2E);  // g-gate: tanh
    bs0[nt] = (bih0[g] + bhh0[g]) * scale;
    bs1[nt] = (bih1[g] + bhh1[g]) * scale;
  }

  const v8f zc = {0.f, 0.f, 0.f, 0.f, 0.f, 0.f, 0.f, 0.f};  // loop-invariant C

  LayerState S0, S1;
  zero_state(S0);
  zero_state(S1);

  // Rotating x prefetch: load for t+1 issued at t, consumed next iteration,
  // hiding VMEM latency behind two full LSTM cell steps.
  float xv = xr[0];
#pragma unroll 1
  for (int t = 0; t < SEQ; t++) {
    int tn = (t + 1 < SEQ) ? (t + 1) : t;
    float xv_next = xr[tn];

    // linear_1 + ReLU folded into the layer-0 A tile (A[m][k] per lane)
    v16h Ain;
#pragma unroll
    for (int e = 0; e < 16; e++) {
      float v = __builtin_fmaf(xv, W1k[e], b1k[e]);
      Ain[e] = (_Float16)fmaxf(v, 0.0f);
    }
    lstm_step(Ain,   Bih0v, Bhh0v, bs0, zc, S0, sh0, m, ksel);
    lstm_step(S0.hA, Bih1v, Bhh1v, bs1, zc, S1, sh1, m, ksel);

    xv = xv_next;
  }

  // Final projection: out[b] = [hf0 | hf1] . W2 + b2, staged through LDS as [16][64]
  float* fin = sh;
#pragma unroll
  for (int r = 0; r < 8; r++) {
    int row = (r + 8 * ksel) * 64;
    fin[row + m]      = S0.hDa[r];
    fin[row + 16 + m] = S0.hDb[r];
    fin[row + 32 + m] = S1.hDa[r];
    fin[row + 48 + m] = S1.hDb[r];
  }
  asm volatile("s_wait_dscnt 0" ::: "memory");
  if (lane < 16) {
    float a = b2[0];
#pragma unroll 8
    for (int j = 0; j < 64; j++) a = __builtin_fmaf(fin[lane * 64 + j], W2[j], a);
    out[tile * 16 + lane] = a;
  }
}

extern "C" void kernel_launch(void* const* d_in, const int* in_sizes, int n_in,
                              void* d_out, int out_size, void* d_ws, size_t ws_size,
                              hipStream_t stream) {
  const float* x    = (const float*)d_in[0];
  const float* W1   = (const float*)d_in[1];
  const float* b1   = (const float*)d_in[2];
  const float* Wih0 = (const float*)d_in[3];
  const float* Whh0 = (const float*)d_in[4];
  const float* bih0 = (const float*)d_in[5];
  const float* bhh0 = (const float*)d_in[6];
  const float* Wih1 = (const float*)d_in[7];
  const float* Whh1 = (const float*)d_in[8];
  const float* bih1 = (const float*)d_in[9];
  const float* bhh1 = (const float*)d_in[10];
  const float* W2   = (const float*)d_in[11];
  const float* b2   = (const float*)d_in[12];
  float* out = (float*)d_out;

  const int Bsz = in_sizes[0] / SEQ;  // input size I == 1
  dim3 grid(Bsz / 16), block(32);
  hipLaunchKernelGGL(lstm2_fused, grid, block, 0, stream,
                     x, W1, b1, Wih0, Whh0, bih0, bhh0,
                     Wih1, Whh1, bih1, bhh1, W2, b2, out);
}
